// LoRAConvsByRandom_cu_78477642433110
// MI455X (gfx1250) — matrix operand order repaired and checked by reading, not compiled, not run on hardware
//
#include <hip/hip_runtime.h>

// ---------------------------------------------------------------------------
// Problem constants (from the reference)
// ---------------------------------------------------------------------------
#define NB      16          // batch
#define IN_CH   128
#define REP_N   90
#define GHOST_N 38
#define NK      11          // outputs per rep channel
#define OUT_N   (REP_N*NK)  // 990
#define KTAPS   25          // 5x5
#define H       56
#define W       56
#define HW      (H*W)       // 3136
#define OH      60          // conv output (pad 4, k=5)
#define OW      60
#define NPOS    (OH*OW)     // 3600
#define NTILES  (NPOS/16)   // 225 -- exact
#define PADW    64          // padded input tile (56 + 2*4)
#define EPS     1e-5f
// SHIFTS[i] = 51//2 - 5*i - 4 = 21 - 5*i

typedef __attribute__((ext_vector_type(16))) _Float16     v16h;
typedef __attribute__((ext_vector_type(8)))  float        v8f;
typedef __attribute__((ext_vector_type(4)))  unsigned int u32x4;
typedef __attribute__((ext_vector_type(8)))  int          i32x8;
typedef __attribute__((ext_vector_type(4)))  int          i32x4;

// ---------------------------------------------------------------------------
// Kernel 0: fold the 4 weight tensors, invert lora2 permutation per channel,
// localize small_idx.
// ---------------------------------------------------------------------------
__global__ __launch_bounds__(256)
void prep_kernel(const float* __restrict__ w0, const float* __restrict__ w1,
                 const float* __restrict__ w2, const float* __restrict__ w3,
                 const int* __restrict__ lora2_idx, const int* __restrict__ small_idx,
                 float* __restrict__ wsum, int* __restrict__ l2shift,
                 int* __restrict__ small_local)
{
    int i = blockIdx.x * 256 + threadIdx.x;
    if (i < OUT_N * KTAPS)
        wsum[i] = w0[i] + w1[i] + w2[i] + w3[i];
    if (i < OUT_N) {                       // i = c*11 + ii (group position)
        int c  = i / NK;
        int ii = i - c * NK;
        int k  = lora2_idx[i] - c * NK;    // out-channel within group getting SHIFTS[ii]
        l2shift[c * NK + k] = 21 - 5 * ii;
    }
    if (i < REP_N)
        small_local[i] = small_idx[i] - i * NK;
}

// ---------------------------------------------------------------------------
// Kernel 0b: build zero-padded f16 input planes (one 64x64 plane per (b,c))
// so the conv kernel can stage its tile with a single TDM DMA.
// ---------------------------------------------------------------------------
__global__ __launch_bounds__(256)
void pad_kernel(const float* __restrict__ inputs, const int* __restrict__ rep,
                _Float16* __restrict__ pad16)
{
    const int bc = blockIdx.x;
    const int b  = bc / REP_N;
    const int c  = bc - b * REP_N;
    const int ch = rep[c];
    const float* src = inputs + ((size_t)b * IN_CH + ch) * HW;
    _Float16* dst = pad16 + (size_t)bc * (PADW * PADW);
    for (int i = threadIdx.x; i < PADW * PADW; i += 256) {
        int iy = i >> 6, ix = i & 63;
        int yy = iy - 4, xx = ix - 4;
        float v = (yy >= 0 && yy < H && xx >= 0 && xx < W) ? src[yy * W + xx] : 0.f;
        dst[i] = (_Float16)v;
    }
}

// ---------------------------------------------------------------------------
// Kernel 1: per-(b, rep-channel) depthwise conv as WMMA GEMM
//   A (16x32 f16)  : 11x25 folded weights for this channel (zero padded)
//   B (32x16 f16)  : im2col patches for 16 output positions
//   D (16x16 f32)  : 11 outputs x 16 positions
// Input tile staged LDS-side by the Tensor Data Mover (TENSOR_LOAD_TO_LDS,
// synchronized with s_wait_tensorcnt); D scattered into l1/l2/sm LDS planes
// with ds_add_f32 atomics; planes flushed once per block.
// ---------------------------------------------------------------------------
__global__ __launch_bounds__(256)
void conv_wmma_kernel(const _Float16* __restrict__ pad16,
                      const float* __restrict__ wsum,
                      const int*   __restrict__ l2shift,
                      const int*   __restrict__ small_local,
                      float* __restrict__ l1g,
                      float* __restrict__ l2g,
                      float* __restrict__ smg)
{
    __shared__ _Float16 sIn[PADW * PADW];   // 8 KB zero-padded input, f16
    __shared__ float sL1[HW];               // 12.25 KB each
    __shared__ float sL2[HW];
    __shared__ float sSm[HW];
    __shared__ int   sShift2[NK];
    __shared__ int   sSmallK;

    const int bc   = blockIdx.x;
    const int b    = bc / REP_N;
    const int c    = bc - b * REP_N;
    const int tid  = threadIdx.x;
    const int lane = tid & 31;
    const int wave = tid >> 5;              // 0..7 (wave32)

    // zero the accumulation planes
    for (int i = tid; i < HW; i += 256) { sL1[i] = 0.f; sL2[i] = 0.f; sSm[i] = 0.f; }
    if (tid < NK)  sShift2[tid] = l2shift[c * NK + tid];
    if (tid == 0)  sSmallK = small_local[c];

    // ---- stage the 64x64 f16 tile into LDS -------------------------------
    const _Float16* src16 = pad16 + (size_t)bc * (PADW * PADW);
#if __has_builtin(__builtin_amdgcn_tensor_load_to_lds)
    if (tid < 32) {   // wave 0 issues one TDM descriptor for the whole tile
        unsigned lds_base = (unsigned)(size_t)(&sIn[0]);
        unsigned long long ga = (unsigned long long)(size_t)src16;
        // D# group0: count=1 | lds_addr | global_addr[56:0] | type=2
        u32x4 g0 = { 1u,
                     lds_base,
                     (unsigned)(ga & 0xFFFFFFFFu),
                     ((unsigned)((ga >> 32) & 0x01FFFFFFu)) | (2u << 30) };
        // D# group1: data_size=2B; tensor 64x64, tile 64x64, dim0 stride 64
        i32x8 g1 = { (int)(1u << 16),       // workgroup_mask=0, data_size=1(2B)
                     (int)(64u << 16),      // tensor_dim0[15:0] at bits[63:48]
                     (int)(64u << 16),      // tensor_dim0 hi=0 | tensor_dim1 lo
                     (int)(64u << 16),      // tensor_dim1 hi=0 | tile_dim0=64
                     64,                    // tile_dim1=64 | tile_dim2=0
                     64,                    // tensor_dim0_stride=64 (lo 32)
                     0, 0 };                // stride hi, tensor_dim1_stride=0
        i32x4 z4 = { 0, 0, 0, 0 };
#if defined(__clang_major__) && (__clang_major__ >= 23)
        i32x8 z8 = { 0, 0, 0, 0, 0, 0, 0, 0 };
        __builtin_amdgcn_tensor_load_to_lds(g0, g1, z4, z4, z8, 0);
#else
        __builtin_amdgcn_tensor_load_to_lds(g0, g1, z4, z4, 0);
#endif
        __builtin_amdgcn_s_wait_tensorcnt(0);
    }
#else
    for (int i = tid; i < PADW * PADW; i += 256) sIn[i] = src16[i];
#endif

    // ---- build A fragment once (ISA 16-bit A 16x32 layout) ----------------
    //  lanes 0-15 : M=lane,    elem j -> K = (j<8 ? j   : j+8)
    //  lanes 16-31: M=lane-16, elem j -> K = (j<8 ? j+8 : j+16)
    v16h afrag;
    {
        const int m = lane & 15;
        #pragma unroll
        for (int j = 0; j < 16; ++j) {
            int k = (lane < 16) ? ((j < 8) ? j : j + 8)
                                : ((j < 8) ? j + 8 : j + 16);
            float wv = 0.f;
            if (m < NK && k < KTAPS)
                wv = wsum[(size_t)(c * NK + m) * KTAPS + k];
            afrag[j] = (_Float16)wv;
        }
    }
    __syncthreads();

    const int n16 = lane & 15;   // this lane's output column within the tile
    const int hi  = lane >> 4;   // D rows 0-7 vs 8-15, B K {0-7,16-23} vs {8-15,24-31}

    // hoist per-lane epilogue metadata out of the tile loop
    bool kval[8], ksm[8];
    int  s2v[8];
    #pragma unroll
    for (int j = 0; j < 8; ++j) {
        const int k = j + (hi << 3);
        kval[j] = (k < NK);
        s2v[j]  = kval[j] ? sShift2[k] : 0;
        ksm[j]  = kval[j] && (k == sSmallK);
    }

    auto do_tile = [&](int t) {
        const int p  = t * 16 + n16;
        const int oy = p / OW;
        const int ox = p - oy * OW;

        // B fragment (ISA 16-bit B 32x16): elem j -> K = 8*hi + j (j<8),
        // 16 + 8*hi + (j-8) (j>=8), matching A.
        v16h bfrag;
        const int kbase = hi << 4;
        #pragma unroll
        for (int j = 0; j < 16; ++j) {
            int kk = (j < 8) ? ((hi << 3) + j) : (16 + (hi << 3) + (j - 8));
            _Float16 v = (_Float16)0.f;
            if (kk < KTAPS) {
                int ky = kk / 5, kx = kk - ky * 5;
                v = sIn[(oy + ky) * PADW + (ox + kx)];
            }
            bfrag[j] = v;
        }

        v8f acc = {};
        acc = __builtin_amdgcn_wmma_f32_16x16x32_f16(
                  false, afrag, false, bfrag, (short)0, acc, false, false);

        // D layout: lane<16 -> acc[j] = out[k=j][p]; lane>=16 -> out[k=j+8][p]
        #pragma unroll
        for (int j = 0; j < 8; ++j) {
            if (kval[j]) {
                const int k = j + (hi << 3);
                const float v = acc[j];
                // l1: shifted along W by SHIFTS[k] = 21-5k
                int y1 = oy - 2, x1 = ox + (21 - 5 * k);
                if ((unsigned)y1 < H && (unsigned)x1 < W)
                    atomicAdd(&sL1[y1 * W + x1], v);            // ds_add_f32
                // l2: shifted along H by per-channel inverse-permuted shift
                int y2 = oy + s2v[j], x2 = ox - 2;
                if ((unsigned)y2 < H && (unsigned)x2 < W)
                    atomicAdd(&sL2[y2 * W + x2], v);
                // sm: center crop of the selected group member
                if (ksm[j]) {
                    int ys = oy - 2, xs = ox - 2;
                    if ((unsigned)ys < H && (unsigned)xs < W)
                        atomicAdd(&sSm[ys * W + xs], v);
                }
            }
        }
    };

    // two independent WMMA tiles per iteration (back-to-back v_wmma, ILP);
    // all control flow is wave-uniform so EXEC is all-ones at every WMMA.
    for (int tp = wave * 2; tp + 1 < NTILES; tp += 16) {
        do_tile(tp);
        do_tile(tp + 1);
    }
    if (wave == 0) do_tile(NTILES - 1);     // tail tile 224

    __syncthreads();

    const size_t base = ((size_t)b * REP_N + c) * HW;
    for (int i = tid; i < HW; i += 256) {
        l1g[base + i] = sL1[i];
        l2g[base + i] = sL2[i];
        smg[base + i] = sSm[i];
    }
}

// ---------------------------------------------------------------------------
// Kernel 2: per-channel 9 moments of (l1, l2, sm) over B*H*W (deterministic
// fixed-order reduction). stats[c*9+..] = S1,S2,S3,Q1,Q2,Q3,P12,P13,P23
// ---------------------------------------------------------------------------
__global__ __launch_bounds__(256)
void stats_kernel(const float* __restrict__ l1g, const float* __restrict__ l2g,
                  const float* __restrict__ smg, float* __restrict__ stats)
{
    __shared__ float red[9 * 256];
    const int c = blockIdx.x;
    const int tid = threadIdx.x;
    float s0=0,s1=0,s2=0,q0=0,q1=0,q2=0,p01=0,p02=0,p12=0;
    for (int n = tid; n < NB * HW; n += 256) {
        int b = n / HW, r = n - b * HW;
        size_t off = ((size_t)b * REP_N + c) * HW + r;
        float a = l1g[off], d = l2g[off], e = smg[off];
        s0 += a; s1 += d; s2 += e;
        q0 += a * a; q1 += d * d; q2 += e * e;
        p01 += a * d; p02 += a * e; p12 += d * e;
    }
    red[0*256+tid]=s0; red[1*256+tid]=s1; red[2*256+tid]=s2;
    red[3*256+tid]=q0; red[4*256+tid]=q1; red[5*256+tid]=q2;
    red[6*256+tid]=p01; red[7*256+tid]=p02; red[8*256+tid]=p12;
    __syncthreads();
    for (int stride = 128; stride > 0; stride >>= 1) {
        if (tid < stride)
            for (int q = 0; q < 9; ++q)
                red[q * 256 + tid] += red[q * 256 + tid + stride];
        __syncthreads();
    }
    if (tid < 9) stats[c * 9 + tid] = red[tid * 256];
}

// ---------------------------------------------------------------------------
// Kernel 3: collapse bn(l1)+bn(l2)+bn(sm) followed by bn_fin into
// y = k1*l1 + k2*l2 + k3*sm + k0 per channel.
//   mean(x) = b1+b2+b3 exactly; var(x) from variances + covariances.
// ---------------------------------------------------------------------------
__global__ void coef_kernel(const float* __restrict__ stats,
                            const float* g1, const float* b1,
                            const float* g2, const float* b2,
                            const float* g3, const float* b3,
                            const float* gf, const float* bf,
                            float* __restrict__ coef)
{
    int c = blockIdx.x * blockDim.x + threadIdx.x;
    if (c >= REP_N) return;
    const float N = (float)(NB * HW);
    const float* s = stats + c * 9;
    float m1 = s[0] / N, m2 = s[1] / N, m3 = s[2] / N;
    float v1 = s[3] / N - m1 * m1, v2 = s[4] / N - m2 * m2, v3 = s[5] / N - m3 * m3;
    float c12 = s[6] / N - m1 * m2, c13 = s[7] / N - m1 * m3, c23 = s[8] / N - m2 * m3;
    float a1 = g1[c] * rsqrtf(v1 + EPS);
    float a2 = g2[c] * rsqrtf(v2 + EPS);
    float a3 = g3[c] * rsqrtf(v3 + EPS);
    float varx = a1*a1*v1 + a2*a2*v2 + a3*a3*v3
               + 2.f * (a1*a2*c12 + a1*a3*c13 + a2*a3*c23);
    float af = gf[c] * rsqrtf(varx + EPS);
    float k1 = af * a1, k2 = af * a2, k3 = af * a3;
    float k0 = bf[c] - (k1 * m1 + k2 * m2 + k3 * m3);
    coef[c*4+0] = k0; coef[c*4+1] = k1; coef[c*4+2] = k2; coef[c*4+3] = k3;
}

// ---------------------------------------------------------------------------
// Kernel 4: fused elementwise output + ghost-channel gather.
// out is (16, 128, 56, 56): channels 0..89 = folded BN result, 90..127 = ghost.
// ---------------------------------------------------------------------------
__global__ __launch_bounds__(256)
void final_kernel(const float* __restrict__ l1g, const float* __restrict__ l2g,
                  const float* __restrict__ smg, const float* __restrict__ coef,
                  const float* __restrict__ inputs, const int* __restrict__ ghost,
                  float* __restrict__ out)
{
    size_t gid = (size_t)blockIdx.x * 256 + threadIdx.x;
    const size_t total = (size_t)NB * IN_CH * HW;
    if (gid >= total) return;
    int r = (int)(gid % HW);
    int c = (int)((gid / HW) % IN_CH);
    int b = (int)(gid / ((size_t)HW * IN_CH));
    float y;
    if (c < REP_N) {
        size_t off = ((size_t)b * REP_N + c) * HW + r;
        const float* k = coef + c * 4;
        y = k[0] + k[1] * l1g[off] + k[2] * l2g[off] + k[3] * smg[off];
    } else {
        int gch = ghost[c - REP_N];
        y = inputs[((size_t)b * IN_CH + gch) * HW + r];
    }
    out[gid] = y;
}

// ---------------------------------------------------------------------------
// Host launcher. Inputs (setup_inputs dict order):
//  0 inputs f32 | 1 ghost i32 | 2 rep i32 | 3 lora2_idx i32 | 4 small_idx i32 |
//  5-8 w0..w3 f32 | 9 g_l1 | 10 b_l1 | 11 g_l2 | 12 b_l2 | 13 g_sm | 14 b_sm |
//  15 g_fin | 16 b_fin
// Workspace (~66 MB): padded f16 planes + 3 f32 planes + metadata.
// ---------------------------------------------------------------------------
extern "C" void kernel_launch(void* const* d_in, const int* in_sizes, int n_in,
                              void* d_out, int out_size, void* d_ws, size_t ws_size,
                              hipStream_t stream)
{
    const float* inputs    = (const float*)d_in[0];
    const int*   ghost     = (const int*)  d_in[1];
    const int*   rep       = (const int*)  d_in[2];
    const int*   lora2_idx = (const int*)  d_in[3];
    const int*   small_idx = (const int*)  d_in[4];
    const float* w0 = (const float*)d_in[5];
    const float* w1 = (const float*)d_in[6];
    const float* w2 = (const float*)d_in[7];
    const float* w3 = (const float*)d_in[8];
    const float* g_l1 = (const float*)d_in[9];
    const float* b_l1 = (const float*)d_in[10];
    const float* g_l2 = (const float*)d_in[11];
    const float* b_l2 = (const float*)d_in[12];
    const float* g_sm = (const float*)d_in[13];
    const float* b_sm = (const float*)d_in[14];
    const float* g_fin = (const float*)d_in[15];
    const float* b_fin = (const float*)d_in[16];
    float* out = (float*)d_out;

    const size_t NL = (size_t)NB * REP_N * HW;   // 4,515,840 floats per plane
    char* wp = (char*)d_ws;
    _Float16* pad16 = (_Float16*)wp;  wp += (size_t)NB * REP_N * PADW * PADW * sizeof(_Float16);
    float* l1g  = (float*)wp;         wp += NL * sizeof(float);
    float* l2g  = (float*)wp;         wp += NL * sizeof(float);
    float* smg  = (float*)wp;         wp += NL * sizeof(float);
    float* wsum = (float*)wp;         wp += (size_t)OUT_N * KTAPS * sizeof(float);
    int*   l2shift     = (int*)wp;    wp += (size_t)OUT_N * sizeof(int);
    int*   small_local = (int*)wp;    wp += (size_t)REP_N * sizeof(int);
    float* stats       = (float*)wp;  wp += (size_t)REP_N * 9 * sizeof(float);
    float* coef        = (float*)wp;

    prep_kernel<<<(OUT_N * KTAPS + 255) / 256, 256, 0, stream>>>(
        w0, w1, w2, w3, lora2_idx, small_idx, wsum, l2shift, small_local);

    pad_kernel<<<NB * REP_N, 256, 0, stream>>>(inputs, rep, pad16);

    conv_wmma_kernel<<<NB * REP_N, 256, 0, stream>>>(
        pad16, wsum, l2shift, small_local, l1g, l2g, smg);

    stats_kernel<<<REP_N, 256, 0, stream>>>(l1g, l2g, smg, stats);

    coef_kernel<<<1, 128, 0, stream>>>(
        stats, g_l1, b_l1, g_l2, b_l2, g_sm, b_sm, g_fin, b_fin, coef);

    const size_t total = (size_t)NB * IN_CH * HW;
    final_kernel<<<(int)((total + 255) / 256), 256, 0, stream>>>(
        l1g, l2g, smg, coef, inputs, ghost, out);
}
